// MinkowskiEncoder_30356828848557
// MI455X (gfx1250) — matrix operand order repaired and checked by reading, not compiled, not run on hardware
//
#include <hip/hip_runtime.h>
#include <hip/hip_bf16.h>

// ---------------- problem constants ----------------
#define BB   4
#define HH   352
#define WW   1216
#define NPTS 52000           // N = B*NPB
#define MM   45000
#define CC   32
#define C2   64
#define K2T  25
#define HD   176
#define WD   608
#define EPSV 1e-5f
#define DENSE_SZ (BB*CC*HD*WD)   // 13,697,024

// ---------------- workspace layout (dword offsets) ----------------
#define OFF_IDXF   0u            // B*H*W            = 1,712,128
#define OFF_IDXH   1712128u      // B*HD*WD          =   428,032
#define OFF_NBR    2140160u      // M*25             = 1,125,000
#define OFF_F      3265160u      // M
#define OFF_XSUM   3310160u      // M*32
#define OFF_PRE2   4750160u      // M*64  (reused for pre3b)
#define OFF_PRE3A  7630160u      // M*64
#define OFF_H2BF   10510160u     // M*64 u16 = 1,440,000 dw (reused for h3b_bf)
#define OFF_H3ABF  11950160u     // 1,440,000 dw
#define OFF_WP2B   13390160u     // 25*64*32 u16 = 25,600 dw
#define OFF_WP3B   13415760u     // 25*64*64 u16 = 51,200 dw
#define OFF_WP3C   13466960u     // 25,600 dw
#define OFF_STATS  13492560u     // 512 f: [s2:128][s3a:128][s3b:128][sF:64]
#define OFF_CHSUM  13493072u     // B*C = 128 f

typedef __attribute__((ext_vector_type(16))) __bf16 bf16x16;
typedef __attribute__((ext_vector_type(8)))  float  f32x8;

union Frag { uint4 u[2]; bf16x16 v; };

__device__ __forceinline__ unsigned short f2bf(float x) {
  unsigned int u = __float_as_uint(x);
  u += 0x7FFFu + ((u >> 16) & 1u);           // round-to-nearest-even
  return (unsigned short)(u >> 16);
}

// ---------------- glue kernels ----------------
__global__ void fill_i32(int* p, int v, int n) {
  int i = blockIdx.x * blockDim.x + threadIdx.x;
  if (i < n) p[i] = v;
}
__global__ void fill_f32(float* p, float v, int n) {
  int i = blockIdx.x * blockDim.x + threadIdx.x;
  if (i < n) p[i] = v;
}
__global__ void scatter_map(const int* __restrict__ coords, int* __restrict__ map,
                            int n, int h, int w) {
  int i = blockIdx.x * blockDim.x + threadIdx.x;
  if (i >= n) return;
  int b = coords[3*i], y = coords[3*i+1], x = coords[3*i+2];
  map[(b*h + y)*w + x] = i;
}
// 3x3 max-pool at full res over the input point cloud
__global__ void maxpool9(const int* __restrict__ cout, const int* __restrict__ idxf,
                         const float* __restrict__ feats, float* __restrict__ f) {
  int n = blockIdx.x * blockDim.x + threadIdx.x;
  if (n >= MM) return;
  int b = cout[3*n], yc = cout[3*n+1]*2, xc = cout[3*n+2]*2;
  float best = -3.0e38f;
  #pragma unroll
  for (int dy = -1; dy <= 1; ++dy)
    #pragma unroll
    for (int dx = -1; dx <= 1; ++dx) {
      int y = yc + dy, x = xc + dx;
      if (y >= 0 && y < HH && x >= 0 && x < WW) {
        int idx = idxf[(b*HH + y)*WW + x];
        if (idx < NPTS) best = fmaxf(best, feats[idx]);
      }
    }
  f[n] = best;
}
// 5x5 neighbor table at half res; invalid -> MM
__global__ void build_nbr25(const int* __restrict__ cout, const int* __restrict__ idxh,
                            int* __restrict__ nbr) {
  int i = blockIdx.x * blockDim.x + threadIdx.x;
  if (i >= MM * K2T) return;
  int n = i / K2T, k = i % K2T;
  int dy = k / 5 - 2, dx = k % 5 - 2;
  int b = cout[3*n], y = cout[3*n+1] + dy, x = cout[3*n+2] + dx;
  int idx = MM;
  if (y >= 0 && y < HD && x >= 0 && x < WD) idx = idxh[(b*HD + y)*WD + x];
  nbr[i] = idx;
}
// 1-channel 25-tap conv: out[n,d] = sum_k f[nbr[n,k]] * w[k*Cout+d]
__global__ void sconv_c1(const float* __restrict__ f, const int* __restrict__ nbr,
                         const float* __restrict__ w, float* __restrict__ out, int Cout) {
  __shared__ float fk[4][K2T];
  int p = blockIdx.x * 4 + threadIdx.y;
  int t = threadIdx.x;
  if (p < MM && t < K2T) {
    int idx = nbr[p*K2T + t];
    fk[threadIdx.y][t] = (idx < MM) ? f[idx] : 0.f;
  }
  __syncthreads();
  if (p >= MM || t >= Cout) return;
  float s = 0.f;
  #pragma unroll
  for (int k = 0; k < K2T; ++k) s += fk[threadIdx.y][k] * w[k*Cout + t];
  out[p*Cout + t] = s;
}
// per-channel sum / sumsq (stats zeroed beforehand)
__global__ void bn_stats(const float* __restrict__ x, float* __restrict__ stats, int Cch) {
  int c = threadIdx.x;
  int r0 = blockIdx.x * 128;
  int r1 = min(r0 + 128, MM);
  float s = 0.f, s2 = 0.f;
  for (int r = r0; r < r1; ++r) { float v = x[r*Cch + c]; s += v; s2 += v*v; }
  atomicAdd(&stats[c], s);
  atomicAdd(&stats[Cch + c], s2);
}
// fused BN + ReLU + f32->bf16 convert
__global__ void bn_relu_bf16(const float* __restrict__ x, const float* __restrict__ stats,
                             const float* __restrict__ g, const float* __restrict__ b,
                             unsigned short* __restrict__ out, int Cch) {
  int i = blockIdx.x * blockDim.x + threadIdx.x;
  if (i >= MM * Cch) return;
  int c = i % Cch;
  float m  = stats[c] * (1.f / MM);
  float vr = stats[Cch + c] * (1.f / MM) - m * m;
  float y  = (x[i] - m) * rsqrtf(vr + EPSV) * g[c] + b[c];
  out[i] = f2bf(fmaxf(y, 0.f));
}
// pack f32 weights (K2 x 64 x Cout) into bf16 B-fragment order
__global__ void pack_w(const float* __restrict__ w, unsigned short* __restrict__ wp,
                       int Cout, int total) {
  int o = blockIdx.x * blockDim.x + threadIdx.x;
  if (o >= total) return;
  int NT = Cout >> 4;
  int j    =  o        & 15;
  int lane = (o >> 4)  & 31;
  int rest =  o >> 9;
  int nt = rest % NT;  rest /= NT;
  int cb = rest & 1;
  int k  = rest >> 1;
  int Kbase = (lane >= 16) ? 8 : 0;
  int c    = cb*32 + Kbase + (j & 7) + ((j >> 3) << 4);
  int ncol = nt*16 + (lane & 15);
  wp[o] = f2bf(w[(k*C2 + c)*Cout + ncol]);
}

// ---------------- WMMA sparse conv: 64 in-channels, NT*16 out-channels ----------------
template <int NT, bool ACC>
__global__ void __launch_bounds__(128)
sconv64_wmma(const unsigned short* __restrict__ xin,  // M x 64 bf16
             const int* __restrict__ nbr,             // M x 25
             const unsigned short* __restrict__ wp,   // packed B fragments
             float* __restrict__ out) {
  const int lane = threadIdx.x;          // 0..31
  const int tile = blockIdx.x * 4 + threadIdx.y;
  const int mrow = lane & 15;
  const int half = lane >> 4;
  const int n    = tile * 16 + mrow;
  const bool rowok = (n < MM);
  const int nclamp = rowok ? n : 0;

  f32x8 zero = {};
  f32x8 acc[NT];
  #pragma unroll
  for (int t = 0; t < NT; ++t) acc[t] = zero;

  const int*  nrow = nbr + nclamp * K2T;
  const uint4* wq  = (const uint4*)wp;

  for (int k = 0; k < K2T; ++k) {
    if (k + 1 < K2T)  // prefetch next tap's B fragments
      __builtin_prefetch(wp + (size_t)(k + 1) * 2 * NT * 32 * 16, 0, 3);
    int idx = nrow[k];
    bool valid = rowok && (idx < MM);
    const uint4* src = (const uint4*)(xin + (size_t)(valid ? idx : 0) * C2);
    #pragma unroll
    for (int cb = 0; cb < 2; ++cb) {
      Frag a;
      a.u[0] = src[cb*4 + half];        // channels cb*32 + half*8 .. +7
      a.u[1] = src[cb*4 + half + 2];    // channels +16
      if (!valid) { a.u[0] = make_uint4(0,0,0,0); a.u[1] = make_uint4(0,0,0,0); }
      #pragma unroll
      for (int t = 0; t < NT; ++t) {
        Frag bfr;
        const uint4* wptr = wq + (size_t)(((k*2 + cb)*NT + t)*32 + lane) * 2;
        bfr.u[0] = wptr[0];
        bfr.u[1] = wptr[1];
        acc[t] = __builtin_amdgcn_wmma_f32_16x16x32_bf16(
            false, a.v, false, bfr.v, (short)0, acc[t], false, false);
      }
    }
  }
  // C/D layout: lane holds column (lane&15), rows r + half*8
  const int Cout = NT * 16;
  #pragma unroll
  for (int t = 0; t < NT; ++t)
    #pragma unroll
    for (int r = 0; r < 8; ++r) {
      int rr = tile * 16 + half * 8 + r;
      if (rr < MM) {
        float* o = out + (size_t)rr * Cout + t*16 + (lane & 15);
        float v = acc[t][r];
        *o = ACC ? (*o + v) : v;
      }
    }
}

// ---------------- final BN + ReLU + scatter + channel sums ----------------
__global__ void final_scatter(const float* __restrict__ xsum, const float* __restrict__ stats,
                              const float* __restrict__ gF, const float* __restrict__ bF,
                              const int* __restrict__ cout, float* __restrict__ dense,
                              float* __restrict__ chsum) {
  int i = blockIdx.x * blockDim.x + threadIdx.x;
  if (i >= MM * CC) return;
  int n = i >> 5, c = i & 31;
  float m  = stats[c] * (1.f / MM);
  float vr = stats[CC + c] * (1.f / MM) - m * m;
  float y  = (xsum[i] - m) * rsqrtf(vr + EPSV) * gF[c] + bF[c];
  y = fmaxf(y, 0.f);
  int b = cout[3*n], yo = cout[3*n+1], xo = cout[3*n+2];
  dense[((size_t)(b*CC + c)*HD + yo)*WD + xo] = y;
  atomicAdd(&chsum[b*CC + c], y);
}
// params = mean(dense) @ wf^T + bf ; split gamma/beta
__global__ void params_k(const float* __restrict__ chsum, const float* __restrict__ wf,
                         const float* __restrict__ bfv, float* __restrict__ outGB) {
  int i = blockIdx.x * blockDim.x + threadIdx.x;
  if (i >= BB * 128) return;
  int b = i >> 7, j = i & 127;
  float s = bfv[j];
  const float inv = 1.f / (float)(HD * WD);
  #pragma unroll
  for (int c = 0; c < CC; ++c) s += (chsum[b*CC + c] * inv) * wf[j*CC + c];
  if (j < 64) outGB[b*64 + j]            = s;   // gamma
  else        outGB[BB*64 + b*64 + j-64] = s;   // beta
}

// ---------------- launcher ----------------
extern "C" void kernel_launch(void* const* d_in, const int* in_sizes, int n_in,
                              void* d_out, int out_size, void* d_ws, size_t ws_size,
                              hipStream_t stream) {
  const int*   coords_in  = (const int*)  d_in[0];
  const float* feats_in   = (const float*)d_in[1];
  const int*   coords_out = (const int*)  d_in[2];
  const float* w1  = (const float*)d_in[3];
  const float* w2a = (const float*)d_in[4];
  const float* w2b = (const float*)d_in[5];
  const float* w3a = (const float*)d_in[6];
  const float* w3b = (const float*)d_in[7];
  const float* w3c = (const float*)d_in[8];
  const float* g2  = (const float*)d_in[9];
  const float* b2  = (const float*)d_in[10];
  const float* g3a = (const float*)d_in[11];
  const float* b3a = (const float*)d_in[12];
  const float* g3b = (const float*)d_in[13];
  const float* b3b = (const float*)d_in[14];
  const float* gF  = (const float*)d_in[15];
  const float* bF  = (const float*)d_in[16];
  const float* wf  = (const float*)d_in[17];
  const float* bfv = (const float*)d_in[18];

  unsigned int* ws = (unsigned int*)d_ws;
  int*   idxf   = (int*)  (ws + OFF_IDXF);
  int*   idxh   = (int*)  (ws + OFF_IDXH);
  int*   nbr    = (int*)  (ws + OFF_NBR);
  float* fbuf   = (float*)(ws + OFF_F);
  float* xsum   = (float*)(ws + OFF_XSUM);
  float* pre2   = (float*)(ws + OFF_PRE2);    // also pre3b
  float* pre3a  = (float*)(ws + OFF_PRE3A);
  unsigned short* h2bf  = (unsigned short*)(ws + OFF_H2BF);   // also h3b_bf
  unsigned short* h3abf = (unsigned short*)(ws + OFF_H3ABF);
  unsigned short* wp2b  = (unsigned short*)(ws + OFF_WP2B);
  unsigned short* wp3b  = (unsigned short*)(ws + OFF_WP3B);
  unsigned short* wp3c  = (unsigned short*)(ws + OFF_WP3C);
  float* stats2 = (float*)(ws + OFF_STATS);
  float* stats3a = stats2 + 128;
  float* stats3b = stats2 + 256;
  float* statsF  = stats2 + 384;
  float* chsum  = (float*)(ws + OFF_CHSUM);
  float* dense  = (float*)d_out;
  float* outGB  = dense + DENSE_SZ;

  const int T = 256;
  #define GR(n) dim3(((n) + T - 1) / T)

  // 1) init maps / stats / dense output
  fill_i32<<<GR(BB*HH*WW), T, 0, stream>>>(idxf, NPTS, BB*HH*WW);
  fill_i32<<<GR(BB*HD*WD), T, 0, stream>>>(idxh, MM,   BB*HD*WD);
  fill_f32<<<GR(640),      T, 0, stream>>>(stats2, 0.f, 640);   // stats + chsum
  fill_f32<<<GR(DENSE_SZ), T, 0, stream>>>(dense, 0.f, DENSE_SZ);

  // 2) scatter index maps
  scatter_map<<<GR(NPTS), T, 0, stream>>>(coords_in,  idxf, NPTS, HH, WW);
  scatter_map<<<GR(MM),   T, 0, stream>>>(coords_out, idxh, MM,   HD, WD);

  // 3) 3x3 max-pool ; 4) 5x5 neighbor table
  maxpool9<<<GR(MM), T, 0, stream>>>(coords_out, idxf, feats_in, fbuf);
  build_nbr25<<<GR(MM*K2T), T, 0, stream>>>(coords_out, idxh, nbr);

  // 5) pack weights to bf16 B-fragment order
  pack_w<<<GR(K2T*C2*CC), T, 0, stream>>>(w2b, wp2b, CC, K2T*C2*CC);
  pack_w<<<GR(K2T*C2*C2), T, 0, stream>>>(w3b, wp3b, C2, K2T*C2*C2);
  pack_w<<<GR(K2T*C2*CC), T, 0, stream>>>(w3c, wp3c, CC, K2T*C2*CC);

  // 6) 1-channel convs
  dim3 bc32(CC, 4), bc64(C2, 4), gc((MM + 3) / 4);
  sconv_c1<<<gc, bc32, 0, stream>>>(fbuf, nbr, w1,  xsum,  CC);   // x1 (fresh)
  sconv_c1<<<gc, bc64, 0, stream>>>(fbuf, nbr, w2a, pre2,  C2);
  sconv_c1<<<gc, bc64, 0, stream>>>(fbuf, nbr, w3a, pre3a, C2);

  // 7) BN+ReLU -> bf16
  dim3 gst((MM + 127) / 128);
  bn_stats<<<gst, dim3(C2), 0, stream>>>(pre2,  stats2,  C2);
  bn_stats<<<gst, dim3(C2), 0, stream>>>(pre3a, stats3a, C2);
  bn_relu_bf16<<<GR(MM*C2), T, 0, stream>>>(pre2,  stats2,  g2,  b2,  h2bf,  C2);
  bn_relu_bf16<<<GR(MM*C2), T, 0, stream>>>(pre3a, stats3a, g3a, b3a, h3abf, C2);

  // 8) WMMA sparse convs
  dim3 wb(32, 4), wg((MM/16 + 4) / 4 + 1);   // 16 rows per wave, 4 waves per block
  sconv64_wmma<2, true ><<<wg, wb, 0, stream>>>(h2bf,  nbr, wp2b, xsum);  // x2 += (64->32)
  sconv64_wmma<4, false><<<wg, wb, 0, stream>>>(h3abf, nbr, wp3b, pre2);  // pre3b (64->64)
  bn_stats<<<gst, dim3(C2), 0, stream>>>(pre2, stats3b, C2);
  bn_relu_bf16<<<GR(MM*C2), T, 0, stream>>>(pre2, stats3b, g3b, b3b, h2bf, C2); // h3b_bf
  sconv64_wmma<2, true ><<<wg, wb, 0, stream>>>(h2bf, nbr, wp3c, xsum);   // x3 += (64->32)

  // 9) final BN + ReLU + scatter + channel sums
  bn_stats<<<gst, dim3(CC), 0, stream>>>(xsum, statsF, CC);
  final_scatter<<<GR(MM*CC), T, 0, stream>>>(xsum, statsF, gF, bF, coords_out,
                                             dense, chsum);
  // 10) gamma/beta
  params_k<<<GR(BB*128), T, 0, stream>>>(chsum, wf, bfv, outGB);
  #undef GR
}
